// PhysNet_padding_ED_mask_78709570666623
// MI455X (gfx1250) — matrix operand order repaired and checked by reading, not compiled, not run on hardware
//
#include <hip/hip_runtime.h>
#include <hip/hip_bf16.h>
#include <math.h>

// ---------------------------------------------------------------------------
// PhysNet forward for gfx1250 (MI455X).
// Heavy convs -> bf16 WMMA implicit GEMM (v_wmma_f32_16x16x32_bf16).
// Kernel templated on (MT = Cout/16 tiles, kD,kH,kW): compile-time divisors,
// fully static WMMA/accumulator loops.
// Software pipeline: double-buffered LDS weight tile (2 x MT*16x32 bf16);
// chunk k+1 is staged (float4 global loads -> packed ds_store_b64) while
// chunk k is consumed (2x ds_load_b128 per tile -> WMMA). One workgroup
// barrier per K-chunk. Activation gathers are split into a batched load
// phase and a convert phase so loads issue back-to-back before any wait.
// All waves run a uniform trip-count loop (no early return) so barriers and
// the EXEC=all-ones WMMA requirement hold.
// BN+bias+activation fused into epilogues. First conv fused with its maxpool
// to kill a 336MB intermediate. Workspace arena: 125,829,120 floats.
// ---------------------------------------------------------------------------

typedef __attribute__((ext_vector_type(16))) __bf16 v16bf;
typedef __attribute__((ext_vector_type(8)))  float  v8f;
typedef __attribute__((ext_vector_type(4)))  unsigned int v4u;

static __device__ __forceinline__ unsigned short f2bfbits(float f) {
  union { float f; unsigned u; } cvt; cvt.f = f;
  const unsigned u = cvt.u + 0x7FFFu + ((cvt.u >> 16) & 1u);  // RNE
  return (unsigned short)(u >> 16);
}

static __device__ __forceinline__ __bf16 f2bf(float f) {
  const unsigned short s = f2bfbits(f);
  __bf16 r;
  __builtin_memcpy(&r, &s, sizeof(r));
  return r;
}

struct ConvArgs {
  const float* in; const float* w; const float* bias;
  const float* g; const float* be; const float* mu; const float* var;
  float* out;
  int N, Cin, D, H, W;            // input dims (NCDHW)
  int Cout;
  int pD, pH, pW;
  int oD, oH, oW;
  int act;                        // 0 none, 1 relu, 2 elu
  int has_bn;
};

// Stage one 32-wide K chunk of weights (MT*16 rows) into LDS as bf16.
// float4 loads (all K used here are multiples of 4; kk%4==0 -> aligned, and
// each quad is entirely in- or out-of-range), packed bf16x2 ds_store_b64.
template <int MT>
static __device__ __forceinline__ void stage_chunk(
    const float* __restrict__ w, int Cout, int K, int k0, int tid,
    unsigned short* dst) {
  constexpr int NQ = (MT * 128 + 255) / 256;   // float4 quads per thread
  float4 sv[NQ];
  #pragma unroll
  for (int qi = 0; qi < NQ; ++qi) {
    const int q     = qi * 256 + tid;
    const int e4    = q << 2;                  // element index in chunk
    const int mrowl = e4 >> 5;
    const int kk    = e4 & 31;
    const int kg    = k0 + kk;
    const bool ok   = (q < MT * 128) && (mrowl < Cout) && (kg + 4 <= K);
    const long addr = ok ? ((long)mrowl * K + kg) : 0;
    sv[qi] = *reinterpret_cast<const float4*>(w + addr);
  }
  #pragma unroll
  for (int qi = 0; qi < NQ; ++qi) {
    const int q     = qi * 256 + tid;
    const int e4    = q << 2;
    const int mrowl = e4 >> 5;
    const int kk    = e4 & 31;
    const int kg    = k0 + kk;
    const bool ok   = (mrowl < Cout) && (kg + 4 <= K);
    const float4 v  = sv[qi];
    uint2 st;
    st.x = (unsigned)f2bfbits(ok ? v.x : 0.f) |
           ((unsigned)f2bfbits(ok ? v.y : 0.f) << 16);
    st.y = (unsigned)f2bfbits(ok ? v.z : 0.f) |
           ((unsigned)f2bfbits(ok ? v.w : 0.f) << 16);
    if (q < MT * 128)
      *reinterpret_cast<uint2*>(&dst[(mrowl << 5) + kk]) = st;
  }
}

// Fused Conv3D (stride 1) + bias + BN + activation via implicit GEMM.
// GEMM view: M = Cout (MT 16-row tiles), N = n*oD*oH*oW, K = Cin*kD*kH*kW.
// One wave: 16 output positions x all Cout tiles; K consumed 32 at a time.
template <int MT, int KD, int KH, int KW>
__global__ __launch_bounds__(256) void conv3d_wmma_kernel(ConvArgs a) {
  constexpr int kHW  = KH * KW;
  constexpr int kVol = KD * kHW;
  __shared__ alignas(16) unsigned short wtile[2][MT * 16 * 32];  // double buf

  const int  tid  = threadIdx.x;
  const int  lane = tid & 31;
  const long wave = (long)blockIdx.x * (blockDim.x >> 5) + (tid >> 5);
  const long P    = (long)a.N * a.oD * a.oH * a.oW;

  const int hi = lane >> 4;                     // 0 or 1 (lane half)
  const int lm = lane & 15;

  // This lane's output column (shared by all accumulator rows).
  const long p = (wave << 4) + lm;
  const bool pvalid = p < P;
  long pc = pvalid ? p : 0;
  const int ow_ = (int)(pc % a.oW); pc /= a.oW;
  const int oh_ = (int)(pc % a.oH); pc /= a.oH;
  const int od_ = (int)(pc % a.oD);
  const int nb  = (int)(pc / a.oD);

  const int  K   = a.Cin * kVol;
  const long HW  = (long)a.H * a.W;
  const long inC = (long)a.D * HW;
  const long inBase = (long)nb * a.Cin * inC;

  v8f acc[MT];
  #pragma unroll
  for (int mt = 0; mt < MT; ++mt) acc[mt] = v8f{};

  // Prologue: stage chunk 0.
  stage_chunk<MT>(a.w, a.Cout, K, 0, tid, wtile[0]);
  __syncthreads();

  const int nIter = (K + 31) >> 5;
  for (int it = 0; it < nIter; ++it) {
    const int k0 = it << 5;
    unsigned short* curt = wtile[it & 1];

    // Pipeline: stage next chunk into the other buffer (uniform branch).
    if (k0 + 32 < K)
      stage_chunk<MT>(a.w, a.Cout, K, k0 + 32, tid, wtile[(it + 1) & 1]);

    // ---- B gather phase (batched, branch-free clamped loads) ----
    // column = p, K(e) = k0 + e + 16*hi  [contiguous k order, same as A]
    float bx[16]; bool bok[16];
    #pragma unroll
    for (int e = 0; e < 16; ++e) {
      const int kg = k0 + e + (hi << 4);
      bool ok = pvalid && (kg < K);
      const int kgc = ok ? kg : 0;
      const int ci = kgc / kVol;   const int r  = kgc % kVol;
      const int kd = r / kHW;      const int r2 = r % kHW;
      const int kh = r2 / KW;      const int kw = r2 % KW;
      const int id = od_ + kd - a.pD;
      const int ih = oh_ + kh - a.pH;
      const int iw = ow_ + kw - a.pW;
      ok = ok && (unsigned)id < (unsigned)a.D && (unsigned)ih < (unsigned)a.H &&
           (unsigned)iw < (unsigned)a.W;
      const int idc = id < 0 ? 0 : (id >= a.D ? a.D - 1 : id);
      const int ihc = ih < 0 ? 0 : (ih >= a.H ? a.H - 1 : ih);
      const int iwc = iw < 0 ? 0 : (iw >= a.W ? a.W - 1 : iw);
      bx[e] = a.in[inBase + (long)ci * inC + (long)idc * HW + (long)ihc * a.W + iwc];
      bok[e] = ok;
    }
    // ---- B convert phase ----
    v16bf B;
    #pragma unroll
    for (int e = 0; e < 16; ++e) B[e] = f2bf(bok[e] ? bx[e] : 0.f);

    // ---- A fragments from LDS + WMMA, one per Cout tile ----
    // element e maps to kk = e + 16*hi -> two ds_load_b128.
    #pragma unroll
    for (int mt = 0; mt < MT; ++mt) {
      const int m = (mt << 4) + lm;
      union { v4u q[2]; v16bf v; } au;
      au.q[0] = *reinterpret_cast<const v4u*>(&curt[(m << 5) + (hi << 4)]);
      au.q[1] = *reinterpret_cast<const v4u*>(&curt[(m << 5) + 8 + (hi << 4)]);
      acc[mt] = __builtin_amdgcn_wmma_f32_16x16x32_bf16(
          false, au.v, false, B, (short)0, acc[mt], false, false);
    }
    __syncthreads();   // curt reads done; next-buffer stores visible
  }

  // Epilogue: bias + BN + activation; D row = vgpr + 8*hi (per 16-row tile).
  const long oHW = (long)a.oH * a.oW;
  const long obase = (long)nb * a.Cout;
  #pragma unroll
  for (int mt = 0; mt < MT; ++mt) {
    #pragma unroll
    for (int r = 0; r < 8; ++r) {
      const int m = (mt << 4) + r + (hi << 3);
      if (m < a.Cout && pvalid) {
        float v = acc[mt][r] + a.bias[m];
        if (a.has_bn) {
          const float inv = a.g[m] * rsqrtf(a.var[m] + 1e-5f);
          v = v * inv + (a.be[m] - a.mu[m] * inv);
        }
        if (a.act == 1)      v = v > 0.f ? v : 0.f;
        else if (a.act == 2) v = v > 0.f ? v : (expf(v) - 1.f);
        a.out[(obase + m) * (a.oD * oHW) + (long)od_ * oHW +
              (long)oh_ * a.oW + ow_] = v;
      }
    }
  }
}

// Fused cb1 (16x3x(1,5,5), pad (0,2,2)) + BN + ReLU + maxpool(1,2,2).
// Kills the (2,16,160,128,128) 336MB intermediate entirely.
__global__ __launch_bounds__(256) void conv1_pool_kernel(
    const float* __restrict__ x, const float* __restrict__ w,
    const float* __restrict__ b, const float* __restrict__ g,
    const float* __restrict__ be, const float* __restrict__ mu,
    const float* __restrict__ var, float* __restrict__ out) {
  const int oH = 64, oW = 64, D = 160, H = 128, W = 128, Cin = 3, Cout = 16;
  long i = (long)blockIdx.x * blockDim.x + threadIdx.x;
  const long total = 2L * Cout * D * oH * oW;
  if (i >= total) return;
  int pw = (int)(i % oW); long t = i / oW;
  int ph = (int)(t % oH); t /= oH;
  int d  = (int)(t % D);  t /= D;
  int c  = (int)(t % Cout);
  int n  = (int)(t / Cout);
  const float inv = g[c] * rsqrtf(var[c] + 1e-5f);
  const float sh  = be[c] - mu[c] * inv;
  const float bias = b[c];
  float mx = -3.4e38f;
  for (int dy = 0; dy < 2; ++dy)
    for (int dx = 0; dx < 2; ++dx) {
      const int oh = 2 * ph + dy, ow = 2 * pw + dx;
      float acc = bias;
      for (int ci = 0; ci < Cin; ++ci)
        for (int kh = 0; kh < 5; ++kh) {
          const int ih = oh + kh - 2;
          if ((unsigned)ih >= (unsigned)H) continue;
          for (int kw = 0; kw < 5; ++kw) {
            const int iw = ow + kw - 2;
            if ((unsigned)iw >= (unsigned)W) continue;
            acc += w[((c * Cin + ci) * 5 + kh) * 5 + kw] *
                   x[(((long)n * Cin + ci) * D + d) * (long)H * W +
                     (long)ih * W + iw];
          }
        }
      float v = acc * inv + sh;
      v = v > 0.f ? v : 0.f;
      mx = v > mx ? v : mx;
    }
  out[i] = mx;
}

// Max pool, stride == window (all uses in this net).
__global__ __launch_bounds__(256) void maxpool3d_kernel(
    const float* __restrict__ in, float* __restrict__ out,
    int N, int C, int D, int H, int W, int wD, int wH, int wW) {
  const int oD = D / wD, oH = H / wH, oW = W / wW;
  long i = (long)blockIdx.x * blockDim.x + threadIdx.x;
  const long total = (long)N * C * oD * oH * oW;
  if (i >= total) return;
  int ow = (int)(i % oW); long t = i / oW;
  int oh = (int)(t % oH); t /= oH;
  int od = (int)(t % oD); t /= oD;
  int c  = (int)(t % C);
  int n  = (int)(t / C);
  const long base = ((long)n * C + c) * D;
  float mx = -3.4e38f;
  for (int kd = 0; kd < wD; ++kd)
    for (int kh = 0; kh < wH; ++kh)
      for (int kw = 0; kw < wW; ++kw) {
        const float v = in[(base + od * wD + kd) * (long)H * W +
                           (long)(oh * wH + kh) * W + (ow * wW + kw)];
        mx = v > mx ? v : mx;
      }
  out[i] = mx;
}

__global__ __launch_bounds__(256) void add_kernel(
    const float* __restrict__ a, const float* __restrict__ b,
    float* __restrict__ out, long n) {
  long i = (long)blockIdx.x * blockDim.x + threadIdx.x;
  if (i < n) out[i] = a[i] + b[i];
}

// avgpool 2x2 over last two dims of (2,40,16,16), then >= 0.5 -> {0,1}.
__global__ __launch_bounds__(256) void avg_thresh_kernel(
    const float* __restrict__ in, float* __restrict__ out) {
  int i = blockIdx.x * blockDim.x + threadIdx.x;
  if (i >= 2 * 40 * 8 * 8) return;
  int x = i % 8;  int t = i / 8;
  int y = t % 8;  t /= 8;
  int d = t % 40; int n = t / 40;
  const long base = (((long)n * 40 + d) * 16 + 2 * y) * 16 + 2 * x;
  const float avg = 0.25f * (in[base] + in[base + 1] + in[base + 16] + in[base + 17]);
  out[i] = (avg >= 0.5f) ? 1.f : 0.f;
}

// h(n,c,t,y,x) *= mask(n,t,y,x); h is (2,64,40,8,8).
__global__ __launch_bounds__(256) void mask_mul_kernel(
    const float* __restrict__ h, const float* __restrict__ m,
    float* __restrict__ out) {
  const int S = 40 * 8 * 8, C = 64;
  long i = (long)blockIdx.x * blockDim.x + threadIdx.x;
  if (i >= 2L * C * S) return;
  int s = (int)(i % S); long t = i / S;
  int n = (int)(t / C);
  out[i] = h[i] * m[(long)n * S + s];
}

// ConvTranspose over time: kernel 4, stride 2 (lhs_dilation 2), pad 2,
// 64->64 channels, 1x1 spatial. y[t] = sum_k wf[co,ci,k]*xd[t+k-2],
// wf[co,ci,k] = w[ci,co,3-k], xd[2i]=x[i]. Then BN + ELU.
__global__ __launch_bounds__(256) void convT_time_kernel(
    const float* __restrict__ x, const float* __restrict__ w,
    const float* __restrict__ b, const float* __restrict__ g,
    const float* __restrict__ be, const float* __restrict__ mu,
    const float* __restrict__ var, float* __restrict__ out,
    int N, int Din, int HW) {
  const int C = 64;
  const int Dout = 2 * Din;
  long i = (long)blockIdx.x * blockDim.x + threadIdx.x;
  const long total = (long)N * C * Dout * HW;
  if (i >= total) return;
  int hw = (int)(i % HW); long t1 = i / HW;
  int tt = (int)(t1 % Dout); t1 /= Dout;
  int co = (int)(t1 % C);
  int n  = (int)(t1 / C);
  float acc = 0.f;
  for (int k = 0; k < 4; ++k) {
    const int src = tt + k - 2;
    if (src < 0 || (src & 1)) continue;
    const int ti = src >> 1;
    if (ti >= Din) continue;
    const int wk = 3 - k;
    for (int ci = 0; ci < C; ++ci)
      acc += w[((long)ci * C + co) * 4 + wk] *
             x[(((long)n * C + ci) * Din + ti) * (long)HW + hw];
  }
  acc += b[co];
  const float inv = g[co] * rsqrtf(var[co] + 1e-5f);
  float v = acc * inv + (be[co] - mu[co] * inv);
  out[i] = v > 0.f ? v : (expf(v) - 1.f);
}

// mean over (H,W): (2,64,160,8,8) -> (2,64,160).
__global__ __launch_bounds__(256) void mean_hw_kernel(
    const float* __restrict__ in, float* __restrict__ out, int NCD, int HW) {
  int i = blockIdx.x * blockDim.x + threadIdx.x;
  if (i >= NCD) return;
  const float* p = in + (long)i * HW;
  float s = 0.f;
  for (int j = 0; j < HW; ++j) s += p[j];
  out[i] = s / (float)HW;
}

// ---------------------------------------------------------------------------

extern "C" void kernel_launch(void* const* d_in, const int* in_sizes, int n_in,
                              void* d_out, int out_size, void* d_ws, size_t ws_size,
                              hipStream_t stream) {
  (void)in_sizes; (void)n_in; (void)out_size; (void)ws_size;
  const float* x = (const float*)d_in[0];
  auto IN = [&](int i) { return (const float*)d_in[i]; };
  float* out = (float*)d_out;
  float* ws  = (float*)d_ws;

  // Input tuple bases: x=0, cb1..cb9 (6 each), sm1, sm2, sr, so(2), up1, up2, cb10(2)
  const int CB1 = 1, CB2 = 7, CB3 = 13, CB4 = 19, CB5 = 25, CB6 = 31, CB7 = 37,
            CB8 = 43, CB9 = 49, SM1 = 55, SM2 = 61, SR = 67, SO = 73,
            UP1 = 75, UP2 = 81, CB10 = 87;

  // d_out slices (flat float offsets, return-tuple order).
  float* o_rppg = out;                  // (2,1,160)            = 320
  float* o_x    = out + 320;            // (2,3,160,128,128)    = 15,728,640
  float* o_xv32 = out + 15728960;       // (2,64,80,32,32)      = 10,485,760
  float* o_xv16 = out + 26214720;       // (2,64,40,16,16)      = 1,310,720
  float* o_skin = out + 27525440;       // (2,40,16,16)         = 20,480
  float* o_bin  = out + 27545920;       // (2,40,8,8)           = 5,120

  auto run_conv = [&](const float* in, float* outp, int pb, bool bn, int act,
                      int N, int Cin, int D, int H, int W, int Cout,
                      int kD, int kH, int kW, int pD, int pH, int pW) {
    ConvArgs a;
    a.in = in; a.w = IN(pb); a.bias = IN(pb + 1);
    a.g  = bn ? IN(pb + 2) : nullptr; a.be  = bn ? IN(pb + 3) : nullptr;
    a.mu = bn ? IN(pb + 4) : nullptr; a.var = bn ? IN(pb + 5) : nullptr;
    a.out = outp;
    a.N = N; a.Cin = Cin; a.D = D; a.H = H; a.W = W;
    a.Cout = Cout;
    a.pD = pD; a.pH = pH; a.pW = pW;
    a.oD = D + 2 * pD - kD + 1;
    a.oH = H + 2 * pH - kH + 1;
    a.oW = W + 2 * pW - kW + 1;
    a.act = act; a.has_bn = bn ? 1 : 0;
    const long P = (long)N * a.oD * a.oH * a.oW;
    const long waves = (P + 15) / 16;           // one wave = all Cout tiles
    const int blocks = (int)((waves + 7) / 8);
    const int mTiles = (Cout + 15) / 16;
    if (kD == 3) {
      if (mTiles == 4) conv3d_wmma_kernel<4, 3, 3, 3><<<blocks, 256, 0, stream>>>(a);
      else             conv3d_wmma_kernel<2, 3, 3, 3><<<blocks, 256, 0, stream>>>(a);
    } else if (kH == 3) {
      if (mTiles == 2) conv3d_wmma_kernel<2, 1, 3, 3><<<blocks, 256, 0, stream>>>(a);
      else             conv3d_wmma_kernel<1, 1, 3, 3><<<blocks, 256, 0, stream>>>(a);
    } else {
      conv3d_wmma_kernel<1, 1, 1, 1><<<blocks, 256, 0, stream>>>(a);
    }
  };
  auto run_pool = [&](const float* in, float* outp, int N, int C, int D, int H,
                      int W, int wD, int wH, int wW) {
    const long total = (long)N * C * (D / wD) * (H / wH) * (W / wW);
    maxpool3d_kernel<<<(int)((total + 255) / 256), 256, 0, stream>>>(
        in, outp, N, C, D, H, W, wD, wH, wW);
  };

  // Workspace arena (125,829,120 floats). Disjoint live ranges verified.
  float* w0 = ws;               // region A, offset 0
  float* wA = ws + 41943040;    // region B, offset 41,943,040

  // 1. cb1 + BN + ReLU + pool(1,2,2): x -> wA  (2,16,160,64,64)
  conv1_pool_kernel<<<(20971520 + 255) / 256, 256, 0, stream>>>(
      x, IN(CB1), IN(CB1 + 1), IN(CB1 + 2), IN(CB1 + 3), IN(CB1 + 4), IN(CB1 + 5), wA);
  // 2. cb2: wA -> w0  (2,32,160,64,64)
  run_conv(wA, w0, CB2, true, 1, 2, 16, 160, 64, 64, 32, 3, 3, 3, 1, 1, 1);
  // 3. cb3 (xv64): w0 -> wA  (2,64,160,64,64)
  run_conv(w0, wA, CB3, true, 1, 2, 32, 160, 64, 64, 64, 3, 3, 3, 1, 1, 1);
  // 4. pool(2,2,2): wA -> w0  (2,64,80,32,32)
  run_pool(wA, w0, 2, 64, 160, 64, 64, 2, 2, 2);
  // 5. cb4: w0 -> ws+10485760
  run_conv(w0, ws + 10485760, CB4, true, 1, 2, 64, 80, 32, 32, 64, 3, 3, 3, 1, 1, 1);
  // 6. cb5 (xv32): -> d_out slice
  run_conv(ws + 10485760, o_xv32, CB5, true, 1, 2, 64, 80, 32, 32, 64, 3, 3, 3, 1, 1, 1);
  // 7. pool(2,2,2): xv32 -> w0  (2,64,40,16,16) = h16 (live through skin + cb6)
  run_pool(o_xv32, w0, 2, 64, 80, 32, 32, 2, 2, 2);
  // 8-11. skin branch
  run_conv(w0, ws + 2097152, SM1, true, 1, 2, 64, 40, 16, 16, 32, 1, 3, 3, 0, 1, 1);
  run_conv(ws + 2097152, ws + 4194304, SM2, true, 1, 2, 32, 40, 16, 16, 16, 1, 3, 3, 0, 1, 1);
  run_conv(w0, ws + 6291456, SR, true, 1, 2, 64, 40, 16, 16, 16, 1, 1, 1, 0, 0, 0);
  add_kernel<<<(327680 + 255) / 256, 256, 0, stream>>>(
      ws + 4194304, ws + 6291456, ws + 8388608, 327680);
  // 12. so conv (16->1, bias 0.5, no BN/act): -> x_skin slice
  run_conv(ws + 8388608, o_skin, SO, false, 0, 2, 16, 40, 16, 16, 1, 1, 3, 3, 0, 1, 1);
  // 13. avgpool 2x2 + threshold -> binary mask slice
  avg_thresh_kernel<<<(5120 + 255) / 256, 256, 0, stream>>>(o_skin, o_bin);
  // 14. cb6: w0 -> ws+2097152
  run_conv(w0, ws + 2097152, CB6, true, 1, 2, 64, 40, 16, 16, 64, 3, 3, 3, 1, 1, 1);
  // 15. cb7 (xv16): -> d_out slice
  run_conv(ws + 2097152, o_xv16, CB7, true, 1, 2, 64, 40, 16, 16, 64, 3, 3, 3, 1, 1, 1);
  // 16. pool(1,2,2): xv16 -> w0  (2,64,40,8,8)
  run_pool(o_xv16, w0, 2, 64, 40, 16, 16, 1, 2, 2);
  // 17-18. cb8, cb9
  run_conv(w0, ws + 1048576, CB8, true, 1, 2, 64, 40, 8, 8, 64, 3, 3, 3, 1, 1, 1);
  run_conv(ws + 1048576, w0, CB9, true, 1, 2, 64, 40, 8, 8, 64, 3, 3, 3, 1, 1, 1);
  // 19. mask multiply: w0 * bin -> ws+1048576
  mask_mul_kernel<<<(327680 + 255) / 256, 256, 0, stream>>>(w0, o_bin, ws + 1048576);
  // 20. up1 (time transposed conv + BN + ELU): D 40->80
  convT_time_kernel<<<(655360 + 255) / 256, 256, 0, stream>>>(
      ws + 1048576, IN(UP1), IN(UP1 + 1), IN(UP1 + 2), IN(UP1 + 3), IN(UP1 + 4),
      IN(UP1 + 5), w0, 2, 40, 64);
  // 21. up2: D 80->160
  convT_time_kernel<<<(1310720 + 255) / 256, 256, 0, stream>>>(
      w0, IN(UP2), IN(UP2 + 1), IN(UP2 + 2), IN(UP2 + 3), IN(UP2 + 4),
      IN(UP2 + 5), ws + 1048576, 2, 80, 64);
  // 22. spatial mean: (2,64,160,8,8) -> (2,64,160)
  mean_hw_kernel<<<(20480 + 255) / 256, 256, 0, stream>>>(
      ws + 1048576, w0, 20480, 64);
  // 23. cb10 (1x1x1, 64->1): -> rPPG
  run_conv(w0, o_rppg, CB10, false, 0, 2, 64, 160, 1, 1, 1, 1, 1, 1, 0, 0, 0);
  // 24. echo input x into d_out
  hipMemcpyAsync(o_x, x, 15728640 * sizeof(float), hipMemcpyDeviceToDevice, stream);
}
